// QLSTMGen224_65481071401899
// MI455X (gfx1250) — matrix operand order repaired and checked by reading, not compiled, not run on hardware
//
#include <hip/hip_runtime.h>
#include <hip/hip_bf16.h>

#define T_STEPS 256
#define BATCH   512
#define NFEAT   8
#define HID     256
#define IN_DIM  4
#define COMB    (IN_DIM + HID)   // 260

typedef __attribute__((ext_vector_type(16))) __bf16 v16bf;
typedef __attribute__((ext_vector_type(8)))  __bf16 v8bf;
typedef __attribute__((ext_vector_type(8)))  float  v8f;
typedef __attribute__((ext_vector_type(4)))  int    v4i;
typedef __attribute__((address_space(1))) v4i* v4i_glob;
typedef __attribute__((address_space(3))) v4i* v4i_lds;

__device__ __forceinline__ float sigmoidf(float x) {
    return 1.0f / (1.0f + __expf(-x));
}

// ---------------- QCNN feature extractor (per-row tiny MLP chain) -------------
template <int OD, int ID>
__device__ __forceinline__ void lin_tanh(const float* __restrict__ W,
                                         const float* __restrict__ b,
                                         const float* x, float* y) {
#pragma unroll
    for (int o = 0; o < OD; ++o) {
        float s = b[o];
#pragma unroll
        for (int j = 0; j < ID; ++j) s += W[o * ID + j] * x[j];
        y[o] = tanhf(s);
    }
}

__global__ void qcnn_kernel(const float* __restrict__ in,
                            const float* __restrict__ fm_w, const float* __restrict__ fm_b,
                            const float* __restrict__ c1_w, const float* __restrict__ c1_b,
                            const float* __restrict__ p1_w, const float* __restrict__ p1_b,
                            const float* __restrict__ c2_w, const float* __restrict__ c2_b,
                            const float* __restrict__ p2_w, const float* __restrict__ p2_b,
                            const float* __restrict__ c3_w, const float* __restrict__ c3_b,
                            float* __restrict__ feats) {
    int tb = blockIdx.x * blockDim.x + threadIdx.x;
    if (tb >= T_STEPS * BATCH) return;
    float x0[8], x1[16], x2[16], x3[12], x4[8], x5[4], x6[4];
#pragma unroll
    for (int j = 0; j < 8; ++j) x0[j] = in[(size_t)tb * 8 + j];
    lin_tanh<16, 8>(fm_w, fm_b, x0, x1);
    lin_tanh<16, 16>(c1_w, c1_b, x1, x2);
    lin_tanh<12, 16>(p1_w, p1_b, x2, x3);
    lin_tanh<8, 12>(c2_w, c2_b, x3, x4);
    lin_tanh<4, 8>(p2_w, p2_b, x4, x5);
    lin_tanh<4, 4>(c3_w, c3_b, x5, x6);
#pragma unroll
    for (int j = 0; j < 4; ++j) feats[(size_t)tb * 4 + j] = x6[j];
}

// ------------- pack recurrent weights: Wh_bf16[gate][k][n] = W[n][4+k] --------
__global__ void pack_wh_kernel(const float* __restrict__ fw, const float* __restrict__ iw,
                               const float* __restrict__ uw, const float* __restrict__ ow,
                               __bf16* __restrict__ whb) {
    int i = blockIdx.x * blockDim.x + threadIdx.x;
    if (i >= 4 * HID * HID) return;
    int gate = i >> 16;
    int rem  = i & 65535;
    int k    = rem >> 8;
    int n    = rem & 255;
    const float* W = (gate == 0) ? fw : (gate == 1) ? iw : (gate == 2) ? uw : ow;
    whb[i] = (__bf16)W[n * COMB + IN_DIM + k];
}

// ------------------------ state init (c=0, h0=0, out=head_b) ------------------
__global__ void init_kernel(float* __restrict__ cstate, __bf16* __restrict__ h0,
                            float* __restrict__ out, const float* __restrict__ head_b) {
    int i = blockIdx.x * blockDim.x + threadIdx.x;
    if (i < BATCH * HID) { cstate[i] = 0.0f; h0[i] = (__bf16)0.0f; }
    if (i < T_STEPS * BATCH) out[i] = head_b[0];
}

// ------ async-stage one K-slice of all 4 gate B-tiles (4KB) into LDS ---------
// 256 threads, each moves 16 bytes via the async global->LDS path (ASYNCcnt).
__device__ __forceinline__ void stage_b(const __bf16* __restrict__ whb,
                                        __bf16* ldsbuf, int k0, int n0, int tid) {
    const int gate = tid >> 6;        // 0..3
    const int rem  = tid & 63;
    const int k    = rem >> 1;        // 0..31
    const int half = rem & 1;         // which 8-element half of the 16-col row
    const __bf16* src = whb + (size_t)gate * HID * HID + (size_t)(k0 + k) * HID
                            + n0 + half * 8;
    __bf16* dst = ldsbuf + gate * 512 + k * 16 + half * 8;
    __builtin_amdgcn_global_load_async_to_lds_b128(
        (v4i_glob)(void*)src, (v4i_lds)(void*)dst, 0, 0);
}

// ---------------- fused LSTM step: 4-gate WMMA GEMM + cell update -------------
// grid: 64 blocks x 256 threads (8 waves). Block covers 128 batches x 16 hidden;
// the 4 gate B-tiles per K-slice are async-staged once into LDS and shared by
// all 8 waves (8x less L2 traffic than per-wave loads).
__global__ void __launch_bounds__(256)
lstm_step_kernel(const __bf16* __restrict__ hprev, __bf16* __restrict__ hnext,
                 float* __restrict__ cstate,
                 const float* __restrict__ feats_t,      // [BATCH][4]
                 const __bf16* __restrict__ whb,         // [4][256][256]
                 const float* __restrict__ fw, const float* __restrict__ iw,
                 const float* __restrict__ uw, const float* __restrict__ ow,
                 const float* __restrict__ fb, const float* __restrict__ ib,
                 const float* __restrict__ ub, const float* __restrict__ ob,
                 const float* __restrict__ head_w,
                 float* __restrict__ out_t)              // [BATCH]
{
    __shared__ __bf16 ldsB[2][4 * 32 * 16];              // double-buffered, 8KB

    const int tid   = threadIdx.x;          // 0..255
    const int lane  = tid & 31;
    const int wave  = tid >> 5;             // 0..7
    const int mblk  = blockIdx.x & 3;       // 4 M-blocks of 128 batches
    const int nt    = blockIdx.x >> 2;      // 16 N-tiles
    const int m0    = mblk * 128 + wave * 16;
    const int n0    = nt * 16;
    const int khalf = lane >> 4;            // A-operand K-half per ISA layout
    const int mrow  = m0 + (lane & 15);     // A-operand M per ISA layout

    v8f aF = {}, aI = {}, aG = {}, aO = {};

    stage_b(whb, &ldsB[0][0], 0, n0, tid);  // prologue: stage K-slice 0

#pragma unroll
    for (int kt = 0; kt < 8; ++kt) {
        const int k0 = kt * 32;
        __builtin_amdgcn_s_wait_asynccnt(0);   // our staged slice has landed
        __syncthreads();                        // ...for every wave in the WG

        // B: lane l = column N=n0+l%16, K = {kh*8..+7, 16+kh*8..+7}, from LDS
        const __bf16* bbase = &ldsB[kt & 1][0];
        v16bf bF, bI, bU, bO;
#pragma unroll
        for (int e = 0; e < 16; ++e) {
            const int bi = (khalf * 8 + (e & 7) + (e >> 3) * 16) * 16 + (lane & 15);
            bF[e] = bbase[0 * 512 + bi];
            bI[e] = bbase[1 * 512 + bi];
            bU[e] = bbase[2 * 512 + bi];
            bO[e] = bbase[3 * 512 + bi];
        }

        // overlap: kick off the next K-slice stage while WMMAs run
        if (kt < 7) stage_b(whb, &ldsB[(kt + 1) & 1][0], k0 + 32, n0, tid);

        // A: 16x32 bf16 tile of h_prev; lane holds K = {kh*8..+7, 16+kh*8..+7}
        const __bf16* arow = hprev + (size_t)mrow * HID + k0 + khalf * 8;
        v8bf alo = *(const v8bf*)(arow);
        v8bf ahi = *(const v8bf*)(arow + 16);
        v16bf a;
#pragma unroll
        for (int e = 0; e < 8; ++e) { a[e] = alo[e]; a[e + 8] = ahi[e]; }

        aF = __builtin_amdgcn_wmma_f32_16x16x32_bf16(false, a, false, bF, (short)0, aF, false, false);
        aI = __builtin_amdgcn_wmma_f32_16x16x32_bf16(false, a, false, bI, (short)0, aI, false, false);
        aG = __builtin_amdgcn_wmma_f32_16x16x32_bf16(false, a, false, bU, (short)0, aG, false, false);
        aO = __builtin_amdgcn_wmma_f32_16x16x32_bf16(false, a, false, bO, (short)0, aO, false, false);
    }

    // Epilogue: per-lane column n; x-part (K=4) + bias + activations + cell update
    const int n = n0 + (lane & 15);
    const float wxF0 = fw[n * COMB + 0], wxF1 = fw[n * COMB + 1], wxF2 = fw[n * COMB + 2], wxF3 = fw[n * COMB + 3];
    const float wxI0 = iw[n * COMB + 0], wxI1 = iw[n * COMB + 1], wxI2 = iw[n * COMB + 2], wxI3 = iw[n * COMB + 3];
    const float wxU0 = uw[n * COMB + 0], wxU1 = uw[n * COMB + 1], wxU2 = uw[n * COMB + 2], wxU3 = uw[n * COMB + 3];
    const float wxO0 = ow[n * COMB + 0], wxO1 = ow[n * COMB + 1], wxO2 = ow[n * COMB + 2], wxO3 = ow[n * COMB + 3];
    const float bF = fb[n], bI = ib[n], bU = ub[n], bO = ob[n];
    const float hw = head_w[n];

#pragma unroll
    for (int r = 0; r < 8; ++r) {
        const int m = m0 + r + 8 * khalf;       // C/D layout: vgpr r, lane half
        const float* xp = feats_t + (size_t)m * IN_DIM;
        const float x0 = xp[0], x1 = xp[1], x2 = xp[2], x3 = xp[3];

        float f  = sigmoidf(aF[r] + bF + x0 * wxF0 + x1 * wxF1 + x2 * wxF2 + x3 * wxF3);
        float ii = sigmoidf(aI[r] + bI + x0 * wxI0 + x1 * wxI1 + x2 * wxI2 + x3 * wxI3);
        float g  = tanhf   (aG[r] + bU + x0 * wxU0 + x1 * wxU1 + x2 * wxU2 + x3 * wxU3);
        float o  = sigmoidf(aO[r] + bO + x0 * wxO0 + x1 * wxO1 + x2 * wxO2 + x3 * wxO3);

        const int ci = m * HID + n;
        float cv = f * cstate[ci] + ii * g;
        cstate[ci] = cv;
        float h = o * tanhf(cv);
        hnext[ci] = (__bf16)h;

        // head partial: reduce over the 16 lanes sharing this m
        float hp = hw * h;
        hp += __shfl_xor(hp, 1);
        hp += __shfl_xor(hp, 2);
        hp += __shfl_xor(hp, 4);
        hp += __shfl_xor(hp, 8);
        if ((lane & 15) == 0) atomicAdd(&out_t[m], hp);
    }
}

// --------------------------------- launcher ----------------------------------
extern "C" void kernel_launch(void* const* d_in, const int* in_sizes, int n_in,
                              void* d_out, int out_size, void* d_ws, size_t ws_size,
                              hipStream_t stream) {
    const float* inputs = (const float*)d_in[0];
    const float* fm_w = (const float*)d_in[1];  const float* fm_b = (const float*)d_in[2];
    const float* c1_w = (const float*)d_in[3];  const float* c1_b = (const float*)d_in[4];
    const float* p1_w = (const float*)d_in[5];  const float* p1_b = (const float*)d_in[6];
    const float* c2_w = (const float*)d_in[7];  const float* c2_b = (const float*)d_in[8];
    const float* p2_w = (const float*)d_in[9];  const float* p2_b = (const float*)d_in[10];
    const float* c3_w = (const float*)d_in[11]; const float* c3_b = (const float*)d_in[12];
    const float* f_w  = (const float*)d_in[13]; const float* f_b  = (const float*)d_in[14];
    const float* i_w  = (const float*)d_in[15]; const float* i_b  = (const float*)d_in[16];
    const float* u_w  = (const float*)d_in[17]; const float* u_b  = (const float*)d_in[18];
    const float* o_w  = (const float*)d_in[19]; const float* o_b  = (const float*)d_in[20];
    const float* head_w = (const float*)d_in[21];
    const float* head_b = (const float*)d_in[22];
    float* out = (float*)d_out;

    // Workspace layout (all 256B-aligned by construction):
    char* ws = (char*)d_ws;
    float*  feats  = (float*)(ws + 0);                       // T*B*4 f32   = 2 MB
    float*  cstate = (float*)(ws + 2097152);                 // B*HID f32   = 512 KB
    __bf16* hA     = (__bf16*)(ws + 2097152 + 524288);       // B*HID bf16  = 256 KB
    __bf16* hB     = (__bf16*)(ws + 2097152 + 524288 + 262144);
    __bf16* whb    = (__bf16*)(ws + 2097152 + 524288 + 524288); // 4*256*256 bf16 = 512 KB

    pack_wh_kernel<<<(4 * HID * HID) / 256, 256, 0, stream>>>(f_w, i_w, u_w, o_w, whb);
    qcnn_kernel<<<(T_STEPS * BATCH) / 128, 128, 0, stream>>>(
        inputs, fm_w, fm_b, c1_w, c1_b, p1_w, p1_b, c2_w, c2_b, p2_w, p2_b, c3_w, c3_b, feats);
    init_kernel<<<(T_STEPS * BATCH) / 256, 256, 0, stream>>>(cstate, hA, out, head_b);

    for (int t = 0; t < T_STEPS; ++t) {
        const __bf16* hin = (t & 1) ? hB : hA;
        __bf16*      hout = (t & 1) ? hA : hB;
        lstm_step_kernel<<<(BATCH / 128) * (HID / 16), 256, 0, stream>>>(
            hin, hout, cstate, feats + (size_t)t * BATCH * IN_DIM, whb,
            f_w, i_w, u_w, o_w, f_b, i_b, u_b, o_b, head_w,
            out + (size_t)t * BATCH);
    }
}
